// GANOKernel_30897994727579
// MI455X (gfx1250) — matrix operand order repaired and checked by reading, not compiled, not run on hardware
//
#include <hip/hip_runtime.h>

typedef __attribute__((ext_vector_type(16))) _Float16 v16h;
typedef __attribute__((ext_vector_type(8)))  _Float16 v8h;
typedef __attribute__((ext_vector_type(8)))  float    v8f;

#define LDIM 64
#define QT 16   // queries per wave tile
#define OT 32   // obs per chunk (== WMMA K)

// ---------- prep kernels ----------
// Aq[q][l] = pos_query[q] . (W1[0:3]+W1[6:9])[:,l]
__global__ __launch_bounds__(64)
void gano_prep_aq(const float* __restrict__ pos_query,
                  const float* __restrict__ W1,
                  float* __restrict__ Aq) {
  int q = blockIdx.x, l = threadIdx.x;
  float a = 0.f;
#pragma unroll
  for (int p = 0; p < 3; ++p)
    a += pos_query[q*3+p] * (W1[p*64+l] + W1[(6+p)*64+l]);
  Aq[q*64+l] = a;
}

// Ao[o][l] = b1[l] + pos_obs[o] . (W1[3:6]-W1[6:9])[:,l]
__global__ __launch_bounds__(64)
void gano_prep_ao(const float* __restrict__ pos_obs,
                  const float* __restrict__ W1,
                  const float* __restrict__ b1,
                  float* __restrict__ Ao) {
  int o = blockIdx.x, l = threadIdx.x;
  float a = b1[l];
#pragma unroll
  for (int p = 0; p < 3; ++p)
    a += pos_obs[o*3+p] * (W1[(3+p)*64+l] - W1[(6+p)*64+l]);
  Ao[o*64+l] = a;
}

// Vt[l][o] = (h_obs @ Wv + bv)[o][l]  stored transposed, f16
__global__ __launch_bounds__(64)
void gano_prep_v(const float* __restrict__ h_obs,
                 const float* __restrict__ Wv,
                 const float* __restrict__ bv,
                 _Float16* __restrict__ Vt, int No) {
  int o = blockIdx.x, l = threadIdx.x;
  float a = bv[l];
#pragma unroll 8
  for (int k = 0; k < 64; ++k)
    a += h_obs[o*64+k] * Wv[k*64+l];
  Vt[(size_t)l*No + o] = (_Float16)a;
}

// ---------- helpers ----------
__device__ inline v16h load_frag16(const _Float16* p0, const _Float16* p1) {
  v8h lo = *(const v8h*)p0;
  v8h hi = *(const v8h*)p1;
  v16h r;
#pragma unroll
  for (int i = 0; i < 8; ++i) { r[i] = lo[i]; r[8+i] = hi[i]; }
  return r;
}

__device__ inline float wave_max(float x) {
#pragma unroll
  for (int off = 16; off >= 1; off >>= 1)
    x = fmaxf(x, __shfl_xor(x, off, 32));
  return x;
}
__device__ inline float wave_sum(float x) {
#pragma unroll
  for (int off = 16; off >= 1; off >>= 1)
    x += __shfl_xor(x, off, 32);
  return x;
}

// ---------- main attention kernel (split-K over obs) ----------
// grid = ntiles * nsplit blocks, 1 wave each. Block handles queries
// [tile*16, tile*16+16) x obs slice [split*span, (split+1)*span).
// Emits partial flash-attention state (m, s, acc) per tile/split.
__global__ __launch_bounds__(32)
void gano_attn(const float* __restrict__ Aq, const float* __restrict__ Ao,
               const _Float16* __restrict__ Vt,
               const float* __restrict__ pos_query, const float* __restrict__ pos_obs,
               const float* __restrict__ W2, const float* __restrict__ b2,
               const int* __restrict__ query_batch, const int* __restrict__ obs_batch,
               const int* __restrict__ obs_mask,
               float* __restrict__ Pacc, float* __restrict__ Pm, float* __restrict__ Ps,
               int No, int nsplit)
{
  __shared__ float    sAq[QT][LDIM];
  __shared__ float    sW2[LDIM];
  __shared__ float    sPq[QT][3];
  __shared__ int      sQb[QT];
  __shared__ _Float16 sP[QT][OT];   // exp-weights tile (A-matrix for WMMA)

  const int lane  = threadIdx.x;
  const int tile  = blockIdx.x / nsplit;
  const int split = blockIdx.x % nsplit;
  const int span  = No / nsplit;
  const int obeg  = split * span;
  const int oendc = obeg + span;
  const int qbase = tile * QT;
  const int half  = lane >> 4;      // half-wave id
  const int nlo   = lane & 15;
  const int akb   = half ? 16 : 0;  // A-frag K base (contiguous K, matches B)
  const int bkb   = half ? 16 : 0;  // B-frag K base (contiguous K, matches A)

  // stage per-tile constants into LDS (coalesced)
  for (int i = lane; i < QT*LDIM; i += 32) ((float*)sAq)[i] = Aq[(size_t)qbase*LDIM + i];
  sW2[lane] = W2[lane]; sW2[lane+32] = W2[lane+32];
  if (lane < QT) sQb[lane] = query_batch[qbase + lane];
  for (int i = lane; i < QT*3; i += 32) ((float*)sPq)[i] = pos_query[qbase*3 + i];
  __syncthreads();

  const float b2v = b2[0];
  float m[QT], s[QT];
#pragma unroll
  for (int q = 0; q < QT; ++q) { m[q] = -__builtin_inff(); s[q] = 0.f; }
  v8f c0 = {}, c1 = {}, c2 = {}, c3 = {};

  for (int obase = obeg; obase < oendc; obase += OT) {
    const int o = obase + lane;       // this lane's obs column

    // prefetch next chunk (global_prefetch_b8): hide L2 latency under S-loop
    if (obase + OT < oendc) {
      __builtin_prefetch(Ao + (size_t)(o + OT) * LDIM, 0, 3);
      __builtin_prefetch(Vt + (size_t)nlo * No + obase + OT + bkb, 0, 3);
      __builtin_prefetch(Vt + (size_t)(32 + nlo) * No + obase + OT + bkb, 0, 3);
    }

    // ---- logits S[q] = relu(Aq[q]+Ao[o]) . W2  (separable MLP trick) ----
    float S[QT];
#pragma unroll
    for (int q = 0; q < QT; ++q) S[q] = 0.f;
    const float4* aoRow = (const float4*)(Ao + (size_t)o * LDIM);
    for (int g = 0; g < LDIM/4; ++g) {
      float4 ao4 = aoRow[g];
      float4 w24 = *(const float4*)&sW2[g*4];
#pragma unroll
      for (int q = 0; q < QT; ++q) {
        float4 aq4 = *(const float4*)&sAq[q][g*4];
        S[q] += fmaxf(aq4.x + ao4.x, 0.f) * w24.x;
        S[q] += fmaxf(aq4.y + ao4.y, 0.f) * w24.y;
        S[q] += fmaxf(aq4.z + ao4.z, 0.f) * w24.z;
        S[q] += fmaxf(aq4.w + ao4.w, 0.f) * w24.w;
      }
    }

    // ---- mask + online softmax ----
    const int   ob  = obs_batch[o];
    const int   om  = obs_mask[o];
    const float pox = pos_obs[o*3+0], poy = pos_obs[o*3+1], poz = pos_obs[o*3+2];
    float scale[QT];
#pragma unroll
    for (int q = 0; q < QT; ++q) {
      float dx = sPq[q][0]-pox, dy = sPq[q][1]-poy, dz = sPq[q][2]-poz;
      float d2 = dx*dx + dy*dy + dz*dz;
      bool valid = (ob == sQb[q]) && (d2 <= 1.0f) && (om != 0);
      float logit = S[q] + b2v;
      float sl = valid ? logit : -__builtin_inff();
      float cm = wave_max(sl);                  // chunk row-max (uniform)
      float mn = fmaxf(m[q], cm);
      float sc, e;
      if (mn == -__builtin_inff()) { sc = 1.0f; e = 0.0f; }     // fully masked so far
      else { sc = __expf(m[q] - mn); e = valid ? __expf(logit - mn) : 0.0f; }
      s[q] = s[q]*sc + wave_sum(e);
      m[q] = mn;
      scale[q] = sc;
      sP[q][lane] = (_Float16)e;                // stage P tile for transpose
    }

    // rescale running accumulators (C layout: lane half selects M=r or M=8+r)
#pragma unroll
    for (int r = 0; r < 8; ++r) {
      float sc = half ? scale[r+8] : scale[r];
      c0[r] *= sc; c1[r] *= sc; c2[r] *= sc; c3[r] *= sc;
    }
    __syncthreads();

    // ---- A fragment: P[16x32] f16, lane row M=nlo, K = akb..akb+15 ----
    v16h a = load_frag16(&sP[nlo][akb], &sP[nlo][akb + 8]);

    // ---- B fragments from Vt (K contiguous per lane), 4 N-tiles of 16 ----
    const _Float16* vb0 = Vt + (size_t)( 0 + nlo)*No + obase + bkb;
    const _Float16* vb1 = Vt + (size_t)(16 + nlo)*No + obase + bkb;
    const _Float16* vb2 = Vt + (size_t)(32 + nlo)*No + obase + bkb;
    const _Float16* vb3 = Vt + (size_t)(48 + nlo)*No + obase + bkb;
    v16h bA = load_frag16(vb0, vb0 + 8);
    v16h bB = load_frag16(vb1, vb1 + 8);
    v16h bC = load_frag16(vb2, vb2 + 8);
    v16h bD = load_frag16(vb3, vb3 + 8);

    c0 = __builtin_amdgcn_wmma_f32_16x16x32_f16(false, a, false, bA, (short)0, c0, false, false);
    c1 = __builtin_amdgcn_wmma_f32_16x16x32_f16(false, a, false, bB, (short)0, c1, false, false);
    c2 = __builtin_amdgcn_wmma_f32_16x16x32_f16(false, a, false, bC, (short)0, c2, false, false);
    c3 = __builtin_amdgcn_wmma_f32_16x16x32_f16(false, a, false, bD, (short)0, c3, false, false);
    __syncthreads();   // protect sP before next chunk overwrites
  }

  // ---- write partial state (no normalization yet) ----
#pragma unroll
  for (int r = 0; r < 8; ++r) {
    int M = r + 8*half;
    size_t row = ((size_t)blockIdx.x * QT + M) * LDIM;
    Pacc[row + 0*16 + nlo] = c0[r];
    Pacc[row + 1*16 + nlo] = c1[r];
    Pacc[row + 2*16 + nlo] = c2[r];
    Pacc[row + 3*16 + nlo] = c3[r];
  }
#pragma unroll
  for (int q = 0; q < QT; ++q) {
    if (lane == q) {                       // const-indexed, exec-masked stores
      Pm[(size_t)blockIdx.x * QT + q] = m[q];
      Ps[(size_t)blockIdx.x * QT + q] = s[q];
    }
  }
}

// ---------- combine split-K partials and normalize ----------
__global__ __launch_bounds__(64)
void gano_combine(const float* __restrict__ Pacc, const float* __restrict__ Pm,
                  const float* __restrict__ Ps, float* __restrict__ out, int nsplit)
{
  const int tile = blockIdx.x;
  const int col  = threadIdx.x;
  for (int q = 0; q < QT; ++q) {
    float mg = -__builtin_inff();
    for (int i = 0; i < nsplit; ++i)
      mg = fmaxf(mg, Pm[(size_t)(tile*nsplit + i) * QT + q]);
    float sg = 0.f, acc = 0.f;
    for (int i = 0; i < nsplit; ++i) {
      float mi = Pm[(size_t)(tile*nsplit + i) * QT + q];
      float w  = (mi == -__builtin_inff()) ? 0.f : __expf(mi - mg);
      sg  += w * Ps[(size_t)(tile*nsplit + i) * QT + q];
      acc += w * Pacc[((size_t)(tile*nsplit + i) * QT + q) * LDIM + col];
    }
    out[((size_t)tile * QT + q) * LDIM + col] = acc / fmaxf(sg, 1e-30f);
  }
}

extern "C" void kernel_launch(void* const* d_in, const int* in_sizes, int n_in,
                              void* d_out, int out_size, void* d_ws, size_t ws_size,
                              hipStream_t stream) {
  const float* h_obs       = (const float*)d_in[0];
  const float* pos_obs     = (const float*)d_in[1];
  const float* pos_query   = (const float*)d_in[2];
  const float* W1          = (const float*)d_in[3];
  const float* b1          = (const float*)d_in[4];
  const float* W2          = (const float*)d_in[5];
  const float* b2          = (const float*)d_in[6];
  const float* Wv          = (const float*)d_in[7];
  const float* bv          = (const float*)d_in[8];
  const int*   obs_mask    = (const int*)d_in[9];
  const int*   obs_batch   = (const int*)d_in[10];
  const int*   query_batch = (const int*)d_in[11];

  const int Nq = in_sizes[2] / 3;   // 1024
  const int No = in_sizes[1] / 3;   // 2048
  const int ntiles = Nq / QT;       // 64

  // workspace layout
  float*    Aq = (float*)d_ws;
  float*    Ao = Aq + (size_t)Nq * LDIM;
  _Float16* Vt = (_Float16*)(Ao + (size_t)No * LDIM);
  float*    Pacc = (float*)(Vt + (size_t)LDIM * No);
  const size_t base = (size_t)(Nq + No) * LDIM * 4 + (size_t)LDIM * No * 2;

  // largest split factor whose partial buffers fit in ws (deterministic)
  int nsplit = 1;
  for (int cand = 8; cand >= 1; cand >>= 1) {
    size_t need = base + (size_t)ntiles * cand * QT * (LDIM + 2) * sizeof(float);
    if ((No / cand) % OT == 0 && need <= ws_size) { nsplit = cand; break; }
  }
  float* Pm = Pacc + (size_t)ntiles * nsplit * QT * LDIM;
  float* Ps = Pm + (size_t)ntiles * nsplit * QT;

  gano_prep_aq<<<Nq, 64, 0, stream>>>(pos_query, W1, Aq);
  gano_prep_ao<<<No, 64, 0, stream>>>(pos_obs, W1, b1, Ao);
  gano_prep_v <<<No, 64, 0, stream>>>(h_obs, Wv, bv, Vt, No);
  gano_attn<<<ntiles * nsplit, 32, 0, stream>>>(Aq, Ao, Vt, pos_query, pos_obs,
                                                W2, b2, query_batch, obs_batch, obs_mask,
                                                Pacc, Pm, Ps, No, nsplit);
  gano_combine<<<ntiles, 64, 0, stream>>>(Pacc, Pm, Ps, (float*)d_out, nsplit);
}
